// HyperGRUCell_70549132804739
// MI455X (gfx1250) — matrix operand order repaired and checked by reading, not compiled, not run on hardware
//
#include <hip/hip_runtime.h>
#include <math.h>

typedef __attribute__((ext_vector_type(16))) _Float16 v16h;
typedef __attribute__((ext_vector_type(8)))  _Float16 v8h;
typedef __attribute__((ext_vector_type(8)))  float    v8f;

#define NDIM 256
#define EPSF 1e-15f
#define MAXNF 0.996f          // (1 - 4e-3)/sqrt(c), c = 1

// ---------------------------------------------------------------------------
// Block-wide reductions (blockDim.x == 256)
// ---------------------------------------------------------------------------
__device__ __forceinline__ float blk_reduce(float v, float* sred) {
  int tid = threadIdx.x;
  __syncthreads();               // protect re-use of sred across calls
  sred[tid] = v;
  __syncthreads();
  for (int off = 128; off > 0; off >>= 1) {
    if (tid < off) sred[tid] += sred[tid + off];
    __syncthreads();
  }
  return sred[0];
}

__device__ __forceinline__ void blk_reduce3(float a, float b, float c,
                                            float* s0, float* s1, float* s2,
                                            float* out) {
  int tid = threadIdx.x;
  __syncthreads();
  s0[tid] = a; s1[tid] = b; s2[tid] = c;
  __syncthreads();
  for (int off = 128; off > 0; off >>= 1) {
    if (tid < off) {
      s0[tid] += s0[tid + off];
      s1[tid] += s1[tid + off];
      s2[tid] += s2[tid + off];
    }
    __syncthreads();
  }
  out[0] = s0[0]; out[1] = s1[0]; out[2] = s2[0];
}

// ---------------------------------------------------------------------------
// K1: per-row prep.  blockIdx.y: 0..3 = z matrices -> f16 zhat rows, ||z||,
// ||a|| = ||z||/cosh^2(r).  4 = hidden, 5 = hyp_x -> ||x||^2 and f16 copy.
// All GEMM operands leave this kernel already in f16 so the WMMA loop has
// no conversion VALU in it.
// ---------------------------------------------------------------------------
__global__ void hgru_prep(const float* __restrict__ z0, const float* __restrict__ z1,
                          const float* __restrict__ z2, const float* __restrict__ z3,
                          const float* __restrict__ r0, const float* __restrict__ r1,
                          const float* __restrict__ r2, const float* __restrict__ r3,
                          const float* __restrict__ hidden, const float* __restrict__ hypx,
                          _Float16* __restrict__ zh16, _Float16* __restrict__ xh16,
                          float* __restrict__ znv, float* __restrict__ nav,
                          float* __restrict__ x2v) {
  __shared__ float sred[256];
  const int row = blockIdx.x, which = blockIdx.y, tid = threadIdx.x;
  if (which < 4) {
    const float* zm = (which == 0) ? z0 : (which == 1) ? z1 : (which == 2) ? z2 : z3;
    const float* rv = (which == 0) ? r0 : (which == 1) ? r1 : (which == 2) ? r2 : r3;
    const float v  = zm[row * NDIM + tid];
    const float ss = blk_reduce(v * v, sred);
    const float zn = sqrtf(ss);
    // a_unit == zhat (cosh^2 factor cancels; F.normalize eps = 1e-12)
    zh16[(which * NDIM + row) * NDIM + tid] = (_Float16)(v / fmaxf(zn, 1e-12f));
    if (tid == 0) {
      znv[which * NDIM + row] = zn;
      const float ch = coshf(rv[row]);     // sc = 1
      nav[which * NDIM + row] = zn / (ch * ch);
    }
  } else {
    const float* x = (which == 4) ? hidden : hypx;
    const float v  = x[row * NDIM + tid];
    xh16[(which - 4) * NDIM * NDIM + row * NDIM + tid] = (_Float16)v;
    const float ss = blk_reduce(v * v, sred);
    if (tid == 0) x2v[(which - 4) * NDIM + row] = ss;
  }
}

// ---------------------------------------------------------------------------
// K2: the 4 Gram GEMMs  G_f[b,cl] = X_f[b,:] . Zhat_f[cl,:]   (256x256x256)
// One wave32 per 16x16 output tile, fully unrolled -> 8 static
// v_wmma_f32_16x16x32_f16, operands fetched as aligned b128 f16 loads.
// Fragment layouts used here (wave32; A and B share contiguous k order):
//   A (16x32 f16): lane<16 -> M=lane, K = kk..kk+15
//                  lane>=16 -> M=l15, K = kk+16..kk+31
//   B (32x16 f16): lane<16 -> N=lane, K = kk..kk+15 ; lane>=16 -> K = kk+16..kk+31
//   C/D (16x16 f32): VGPR i, lane<16 -> (M=i, N=lane); lane>=16 -> (M=i+8, N=l15)
// ---------------------------------------------------------------------------
__global__ void hgru_gemm(const _Float16* __restrict__ xh16,
                          const _Float16* __restrict__ zh16,
                          float* __restrict__ G) {
  const int f = blockIdx.y;                                  // which fc
  const _Float16* X  = xh16 + ((f & 1) ? NDIM * NDIM : 0);   // f0,f2: hidden; f1,f3: hyp_x
  const _Float16* Zh = zh16 + f * NDIM * NDIM;
  float* Gm          = G    + f * NDIM * NDIM;

  const int lane = threadIdx.x & 31;
  const int wv   = threadIdx.x >> 5;                  // 8 waves / block
  const int t    = blockIdx.x * 8 + wv;               // 256 tiles per GEMM
  const int tm   = t >> 4, tn = t & 15;
  const int half = lane >> 4, l15 = lane & 15;

  const _Float16* arow = X  + (tm * 16 + l15) * NDIM + half * 16;
  const _Float16* brow = Zh + (tn * 16 + l15) * NDIM + half * 16;

  v8f acc = {};
#pragma unroll
  for (int kk = 0; kk < NDIM; kk += 32) {
    const v8h alo = *(const v8h*)(arow + kk);         // K = kk+half*16  .. +7
    const v8h ahi = *(const v8h*)(arow + kk + 8);     // K = kk+half*16+8.. +7
    const v16h b  = *(const v16h*)(brow + kk);        // K = kk+half*16  .. +15
    v16h a;
#pragma unroll
    for (int i = 0; i < 8; ++i) { a[i] = alo[i]; a[i + 8] = ahi[i]; }
    acc = __builtin_amdgcn_wmma_f32_16x16x32_f16(
        /*neg_a=*/false, a, /*neg_b=*/false, b,
        /*c_mod=*/(short)0, acc, /*reuse_a=*/false, /*reuse_b=*/false);
  }
#pragma unroll
  for (int i = 0; i < 8; ++i) {
    Gm[(tm * 16 + half * 8 + i) * NDIM + tn * 16 + l15] = acc[i];
  }
}

// ---------------------------------------------------------------------------
// K3: elementwise _mlr scalar chain + sinh, plus per-row sum(w^2) for the
// _fc scale.  One block per b (row of logits.T), threads over cl.
// Uses the rank-1 structure: q = alpha * zhat, a_unit = zhat, so every dot
// product is closed-form in (G, ||x||^2, ||z||, r).
// ---------------------------------------------------------------------------
__global__ void hgru_mlr(const float* __restrict__ G, const float* __restrict__ znv,
                         const float* __restrict__ nav, const float* __restrict__ x2v,
                         const float* __restrict__ r0, const float* __restrict__ r1,
                         const float* __restrict__ r2, const float* __restrict__ r3,
                         float* __restrict__ Wm, float* __restrict__ scalev) {
  __shared__ float sred[256];
  const int b = blockIdx.x, cl = threadIdx.x;
#pragma unroll
  for (int f = 0; f < 4; ++f) {
    const float* rv = (f == 0) ? r0 : (f == 1) ? r1 : (f == 2) ? r2 : r3;
    const float rb = rv[b];
    const float zn = znv[f * NDIM + cl];
    const float na = nav[f * NDIM + cl];
    const float x2 = x2v[(f & 1) * NDIM + b];          // hidden or hyp_x row norm^2
    const float Gv = G[f * NDIM * NDIM + b * NDIM + cl];

    // q = expmap0(r_b * z_cl) = alpha * zhat  (with geoopt ball projection)
    const float n  = fmaxf(fabsf(rb) * zn, EPSF);
    const float g  = tanhf(n) / n;
    float alpha    = g * rb * zn;
    const float qn = fmaxf(fabsf(alpha), EPSF);
    if (qn > MAXNF) alpha *= MAXNF / qn;
    const float q2 = alpha * alpha;

    // px = project(mobius_add(-q, x_b)) = (numA*zhat + numB*x_b) * s / den
    const float xy   = -alpha * Gv;                    // (-q) . x_b
    const float den  = fmaxf(1.0f + 2.0f * xy + q2 * x2, EPSF);
    const float numA = -(1.0f + 2.0f * xy + x2) * alpha;
    const float numB = 1.0f - q2;
    const float nn2  = numA * numA + 2.0f * numA * numB * Gv + numB * numB * x2;
    const float pn   = sqrtf(fmaxf(nn2, 0.0f)) / den;
    float s = 1.0f;
    const float pnc = fmaxf(pn, EPSF);
    if (pnc > MAXNF) s = MAXNF / pnc;
    const float sp  = s * pn;
    const float lam = 2.0f / (1.0f - sp * sp);
    const float dv  = s * (numA + numB * Gv) / den;    // px . a_unit

    const float logit = 2.0f * na * asinhf(dv * lam);
    const float w = sinhf(logit);                      // sc = 1
    Wm[f * NDIM * NDIM + b * NDIM + cl] = w;
    const float ssum = blk_reduce(w * w, sred);
    if (cl == 0) scalev[f * NDIM + b] = sqrtf(ssum) + 1.0f;
  }
}

// ---------------------------------------------------------------------------
// K4: the GRU cell body.  One block per batch row; rowwise mobius ops with
// LDS 3-way dot-product reductions; projection norms computed closed-form.
// ---------------------------------------------------------------------------
__device__ __forceinline__ float mobius_add_row(float xv, float yv,
                                                float* s0, float* s1, float* s2) {
  float rr[3];
  blk_reduce3(xv * xv, yv * yv, xv * yv, s0, s1, s2, rr);
  const float x2 = rr[0], y2 = rr[1], xy = rr[2];
  const float den = fmaxf(1.0f + 2.0f * xy + x2 * y2, EPSF);
  const float c1 = (1.0f + 2.0f * xy + y2) / den;
  const float c2 = (1.0f - x2) / den;
  float res = c1 * xv + c2 * yv;
  const float n2 = c1 * c1 * x2 + 2.0f * c1 * c2 * xy + c2 * c2 * y2;
  const float n = fmaxf(sqrtf(fmaxf(n2, 0.0f)), EPSF);
  if (n > MAXNF) res *= MAXNF / n;
  return res;
}

__device__ __forceinline__ float sig_logmap0_row(float yv,
                                                 float* s0, float* s1, float* s2) {
  float rr[3];
  blk_reduce3(yv * yv, 0.0f, 0.0f, s0, s1, s2, rr);
  const float n = fmaxf(sqrtf(rr[0]), EPSF);
  const float t = atanhf(fminf(n, 1.0f - 1e-7f));
  const float lv = t * yv / n;
  return 1.0f / (1.0f + expf(-lv));
}

__device__ __forceinline__ float pw_mul_row(float wv, float xv,
                                            float* s0, float* s1, float* s2) {
  const float wx = wv * xv;
  float rr[3];
  blk_reduce3(xv * xv, wx * wx, 0.0f, s0, s1, s2, rr);
  const float xn  = fmaxf(sqrtf(rr[0]), EPSF);
  const float wxn = fmaxf(sqrtf(rr[1]), EPSF);
  const float t   = atanhf(fminf(xn, 1.0f - 1e-7f));
  const float val = tanhf(wxn / xn * t);
  float res = val * wx / wxn;
  const float n = fmaxf(fabsf(val), EPSF);             // ||res|| = |tanh(...)|
  if (n > MAXNF) res *= MAXNF / n;
  return res;
}

__global__ void hgru_cell(const float* __restrict__ Wm, const float* __restrict__ scalev,
                          const float* __restrict__ hidden,
                          const float* __restrict__ bz, const float* __restrict__ br,
                          const float* __restrict__ bh, float* __restrict__ out) {
  __shared__ float s0[256], s1[256], s2[256];
  const int b = blockIdx.x, i = threadIdx.x;

  // _fc outputs: note the reference's broadcast quirk — scale indexed by COLUMN.
  const float f0 = Wm[0 * NDIM * NDIM + b * NDIM + i] * scalev[0 * NDIM + i];
  const float f1 = Wm[1 * NDIM * NDIM + b * NDIM + i] * scalev[1 * NDIM + i];
  const float f2 = Wm[2 * NDIM * NDIM + b * NDIM + i] * scalev[2 * NDIM + i];
  const float f3 = Wm[3 * NDIM * NDIM + b * NDIM + i] * scalev[3 * NDIM + i];

  float m = mobius_add_row(f0, f1, s0, s1, s2);
  m = mobius_add_row(m, bz[i], s0, s1, s2);
  const float zg = sig_logmap0_row(m, s0, s1, s2);

  float mr = mobius_add_row(f2, f3, s0, s1, s2);
  mr = mobius_add_row(mr, br[i], s0, s1, s2);
  const float rg = sig_logmap0_row(mr, s0, s1, s2);

  const float ht  = mobius_add_row(rg, bh[i], s0, s1, s2);
  const float hid = hidden[b * NDIM + i];
  const float mh  = mobius_add_row(-hid, ht, s0, s1, s2);
  const float pw  = pw_mul_row(mh, zg, s0, s1, s2);
  out[b * NDIM + i] = mobius_add_row(hid, pw, s0, s1, s2);
}

// ---------------------------------------------------------------------------
extern "C" void kernel_launch(void* const* d_in, const int* in_sizes, int n_in,
                              void* d_out, int out_size, void* d_ws, size_t ws_size,
                              hipStream_t stream) {
  (void)in_sizes; (void)n_in; (void)out_size; (void)ws_size;
  const float* hypx   = (const float*)d_in[0];
  const float* hidden = (const float*)d_in[1];
  const float* Wz_z = (const float*)d_in[2];  const float* Wz_r = (const float*)d_in[3];
  const float* Uz_z = (const float*)d_in[4];  const float* Uz_r = (const float*)d_in[5];
  const float* Wr_z = (const float*)d_in[6];  const float* Wr_r = (const float*)d_in[7];
  const float* Ur_z = (const float*)d_in[8];  const float* Ur_r = (const float*)d_in[9];
  // d_in[10..13] (Wh_*, Uh_*) feed only the dead value in the reference — skipped.
  const float* bz = (const float*)d_in[14];
  const float* br = (const float*)d_in[15];
  const float* bh = (const float*)d_in[16];

  // -------- workspace layout (f32 region, then 32B-aligned f16 region) -----
  float* ws     = (float*)d_ws;
  float* G      = ws;                              // 4 * 65536 f32
  float* Wm     = ws + 4 * NDIM * NDIM;            // 4 * 65536 f32
  float* znv    = ws + 8 * NDIM * NDIM;            // 1024
  float* nav    = znv + 4 * NDIM;                  // 1024
  float* x2v    = nav + 4 * NDIM;                  // 512
  float* scalev = x2v + 2 * NDIM;                  // 1024
  _Float16* zh16 = (_Float16*)(scalev + 4 * NDIM); // 4 * 65536 f16 (32B aligned)
  _Float16* xh16 = zh16 + 4 * NDIM * NDIM;         // 2 * 65536 f16
  // total ~2.8 MB of workspace

  const dim3 blk(256);
  hgru_prep<<<dim3(256, 6), blk, 0, stream>>>(Wz_z, Uz_z, Wr_z, Ur_z,
                                              Wz_r, Uz_r, Wr_r, Ur_r,
                                              hidden, hypx,
                                              zh16, xh16, znv, nav, x2v);
  hgru_gemm<<<dim3(32, 4), blk, 0, stream>>>(xh16, zh16, G);
  hgru_mlr<<<dim3(256), blk, 0, stream>>>(G, znv, nav, x2v,
                                          Wz_r, Uz_r, Wr_r, Ur_r, Wm, scalev);
  hgru_cell<<<dim3(256), blk, 0, stream>>>(Wm, scalev, hidden, bz, br, bh,
                                           (float*)d_out);
}
